// ElasticGNN_37391985279490
// MI455X (gfx1250) — matrix operand order repaired and checked by reading, not compiled, not run on hardware
//
#include <hip/hip_runtime.h>
#include <hip/hip_bf16.h>
#include <math.h>

typedef __attribute__((ext_vector_type(16))) __bf16 v16bf;
typedef __attribute__((ext_vector_type(8)))  __bf16 v8bf;
typedef __attribute__((ext_vector_type(8)))  float  v8f;

#define F_IN    500
#define KPAD    512
#define HID     256
#define NCLS    40
#define NCLSPAD 48
#define GAMMA_C 0.25f
#define OMG_C   0.75f
#define BETA_C  2.0f
#define LAM1_C  3.0f

union AFrag { v16bf v; __bf16 e[16]; };
union CFrag { v8f   v; float  e[8];  };

// ---------------------------------------------------------------- utilities
__global__ void k_zero_f32(float* __restrict__ p, size_t n) {
    size_t i = (size_t)blockIdx.x * blockDim.x + threadIdx.x;
    if (i < n) p[i] = 0.0f;
}
__global__ void k_zero_u32(unsigned* __restrict__ p, size_t n) {
    size_t i = (size_t)blockIdx.x * blockDim.x + threadIdx.x;
    if (i < n) p[i] = 0u;
}

// ------------------------------------------------------- weight transposes
// w1t[n][k] = W1[k][n], bf16, zero-padded to K=512
__global__ void k_prep_w1t(const float* __restrict__ W1, __bf16* __restrict__ w1t) {
    int i = blockIdx.x * blockDim.x + threadIdx.x;
    if (i >= HID * KPAD) return;
    int n = i / KPAD, k = i - n * KPAD;
    float v = (k < F_IN) ? W1[(size_t)k * HID + n] : 0.0f;
    w1t[i] = (__bf16)v;
}
// w2t[n][k] = W2[k][n], bf16, N zero-padded to 48
__global__ void k_prep_w2t(const float* __restrict__ W2, __bf16* __restrict__ w2t) {
    int i = blockIdx.x * blockDim.x + threadIdx.x;
    if (i >= NCLSPAD * HID) return;
    int n = i / HID, k = i - n * HID;
    float v = (n < NCLS) ? W2[(size_t)k * NCLS + n] : 0.0f;
    w2t[i] = (__bf16)v;
}

// ----------------------------------------------------------------- degrees
__global__ void k_count(const int* __restrict__ src, const int* __restrict__ dst,
                        unsigned* __restrict__ cnt, int E) {
    int e = blockIdx.x * blockDim.x + threadIdx.x;
    if (e >= E) return;
    atomicAdd(&cnt[src[e]], 1u);
    atomicAdd(&cnt[dst[e]], 1u);
}
__global__ void k_dis(const unsigned* __restrict__ cnt, float* __restrict__ dis, int N) {
    int n = blockIdx.x * blockDim.x + threadIdx.x;
    if (n >= N) return;
    dis[n] = rsqrtf((float)cnt[n] + 1.0f);
}

// ------------------------------------------------------------ GEMM1 (WMMA)
// h = relu(x[N,500] @ W1[500,256]) -> bf16 [N,256]
// 8 waves/block; wave w owns cols [32w, 32w+31]; block owns 16 rows.
__global__ __launch_bounds__(256) void k_gemm1(const float* __restrict__ x,
                                               const __bf16* __restrict__ w1t,
                                               __bf16* __restrict__ hbf) {
    const int wave = threadIdx.x >> 5;
    const int lane = threadIdx.x & 31;
    const int hi   = lane >> 4;
    const int lm   = lane & 15;
    const int kg   = hi * 16;                 // A K-group offset
    const int mBase = blockIdx.x * 16;

    const float*  arow  = x   + (size_t)(mBase + lm) * F_IN;
    const __bf16* b0row = w1t + (size_t)(wave * 32 + lm) * KPAD;
    const __bf16* b1row = b0row + (size_t)16 * KPAD;

    CFrag acc0, acc1;
    acc0.v = (v8f){0,0,0,0,0,0,0,0};
    acc1.v = (v8f){0,0,0,0,0,0,0,0};

    for (int k0 = 0; k0 < 480; k0 += 32) {
        AFrag a;
        const float4 p0 = *(const float4*)(arow + k0 + kg);
        const float4 p1 = *(const float4*)(arow + k0 + kg + 4);
        const float4 q0 = *(const float4*)(arow + k0 + kg + 8);
        const float4 q1 = *(const float4*)(arow + k0 + kg + 12);
        a.e[0]=(__bf16)p0.x; a.e[1]=(__bf16)p0.y; a.e[2]=(__bf16)p0.z; a.e[3]=(__bf16)p0.w;
        a.e[4]=(__bf16)p1.x; a.e[5]=(__bf16)p1.y; a.e[6]=(__bf16)p1.z; a.e[7]=(__bf16)p1.w;
        a.e[8]=(__bf16)q0.x; a.e[9]=(__bf16)q0.y; a.e[10]=(__bf16)q0.z; a.e[11]=(__bf16)q0.w;
        a.e[12]=(__bf16)q1.x; a.e[13]=(__bf16)q1.y; a.e[14]=(__bf16)q1.z; a.e[15]=(__bf16)q1.w;
        // B 32x16: lane holds col lm, elements 0..15 -> K = k0 + 16*hi + 0..15
        v16bf b0 = *(const v16bf*)(b0row + k0 + hi * 16);
        v16bf b1 = *(const v16bf*)(b1row + k0 + hi * 16);
        acc0.v = __builtin_amdgcn_wmma_f32_16x16x32_bf16(false, a.v, false, b0,
                                                         (short)0, acc0.v, false, false);
        acc1.v = __builtin_amdgcn_wmma_f32_16x16x32_bf16(false, a.v, false, b1,
                                                         (short)0, acc1.v, false, false);
    }
    { // tail: k0 = 480, guard reads beyond F_IN (W1T padding zeroes the products)
        AFrag a;
        #pragma unroll
        for (int j = 0; j < 8; ++j) {
            int k  = 480 + kg + j;
            int k2 = 488 + kg + j;
            a.e[j]     = (__bf16)((k  < F_IN) ? arow[k]  : 0.0f);
            a.e[8 + j] = (__bf16)((k2 < F_IN) ? arow[k2] : 0.0f);
        }
        v16bf b0 = *(const v16bf*)(b0row + 480 + hi * 16);
        v16bf b1 = *(const v16bf*)(b1row + 480 + hi * 16);
        acc0.v = __builtin_amdgcn_wmma_f32_16x16x32_bf16(false, a.v, false, b0,
                                                         (short)0, acc0.v, false, false);
        acc1.v = __builtin_amdgcn_wmma_f32_16x16x32_bf16(false, a.v, false, b1,
                                                         (short)0, acc1.v, false, false);
    }
    #pragma unroll
    for (int r = 0; r < 8; ++r) {
        size_t row = (size_t)(mBase + 8 * hi + r);
        hbf[row * HID + wave * 32 + lm]      = (__bf16)fmaxf(acc0.e[r], 0.0f);
        hbf[row * HID + wave * 32 + 16 + lm] = (__bf16)fmaxf(acc1.e[r], 0.0f);
    }
}

// ------------------------------------------------------------ GEMM2 (WMMA)
// h2 = hbf[N,256] @ W2[256,40] -> f32, written to hh and xc. 3 waves/block.
__global__ __launch_bounds__(96) void k_gemm2(const __bf16* __restrict__ hbf,
                                              const __bf16* __restrict__ w2t,
                                              float* __restrict__ hh,
                                              float* __restrict__ xc) {
    const int wave = threadIdx.x >> 5;        // col tile 0..2
    const int lane = threadIdx.x & 31;
    const int hi   = lane >> 4;
    const int lm   = lane & 15;
    const int kg   = hi * 16;
    const int mBase = blockIdx.x * 16;

    const __bf16* arow = hbf + (size_t)(mBase + lm) * HID;
    const __bf16* brow = w2t + (size_t)(wave * 16 + lm) * HID;

    CFrag acc;
    acc.v = (v8f){0,0,0,0,0,0,0,0};

    #pragma unroll
    for (int k0 = 0; k0 < HID; k0 += 32) {
        AFrag a;
        *(v8bf*)&a.e[0] = *(const v8bf*)(arow + k0 + kg);
        *(v8bf*)&a.e[8] = *(const v8bf*)(arow + k0 + kg + 8);
        v16bf b = *(const v16bf*)(brow + k0 + hi * 16);
        acc.v = __builtin_amdgcn_wmma_f32_16x16x32_bf16(false, a.v, false, b,
                                                        (short)0, acc.v, false, false);
    }
    const int col = wave * 16 + lm;
    if (col < NCLS) {
        #pragma unroll
        for (int r = 0; r < 8; ++r) {
            size_t o = (size_t)(mBase + 8 * hi + r) * NCLS + col;
            hh[o] = acc.e[r];
            xc[o] = acc.e[r];
        }
    }
}

// ------------------------------------------------------- sparse iterations
// y = gamma*hh + (1-gamma)*dis^2*xc  (diagonal/self part of A_hat)
__global__ void k_spmm_init(float* __restrict__ y, const float* __restrict__ hh,
                            const float* __restrict__ xc, const float* __restrict__ dis,
                            size_t total) {
    size_t i = (size_t)blockIdx.x * blockDim.x + threadIdx.x;
    if (i >= total) return;
    int n = (int)(i / NCLS);
    float d = dis[n];
    y[i] = GAMMA_C * hh[i] + OMG_C * d * d * xc[i];
}

// y[s] += (1-g)*dis_s*dis_d*xc[d]; y[d] += (1-g)*dis_s*dis_d*xc[s]
// gridDim = (ceil(E/256), 10): blockIdx.y = channel quad (no int division)
__global__ void k_spmm_edges(float* __restrict__ y, const float* __restrict__ xc,
                             const float* __restrict__ dis,
                             const int* __restrict__ src, const int* __restrict__ dst,
                             int E) {
    int e = blockIdx.x * blockDim.x + threadIdx.x;
    if (e >= E) return;
    int q = blockIdx.y;
    int s = src[e], d = dst[e];
    float ev = OMG_C * dis[s] * dis[d];
    const float4 vs = *(const float4*)(xc + (size_t)s * NCLS + q * 4);
    const float4 vd = *(const float4*)(xc + (size_t)d * NCLS + q * 4);
    float* ys = y + (size_t)s * NCLS + q * 4;
    float* yd = y + (size_t)d * NCLS + q * 4;
    atomicAdd(ys + 0, ev * vd.x); atomicAdd(ys + 1, ev * vd.y);
    atomicAdd(ys + 2, ev * vd.z); atomicAdd(ys + 3, ev * vd.w);
    atomicAdd(yd + 0, ev * vs.x); atomicAdd(yd + 1, ev * vs.y);
    atomicAdd(yd + 2, ev * vs.z); atomicAdd(yd + 3, ev * vs.w);
}

// vs = dis * (y - gamma*dis*t)   (= dis * x_bar, ready for inc())
__global__ void k_compute_vs(float* __restrict__ vs, const float* __restrict__ y,
                             const float* __restrict__ t, const float* __restrict__ dis,
                             size_t total) {
    size_t i = (size_t)blockIdx.x * blockDim.x + threadIdx.x;
    if (i >= total) return;
    int n = (int)(i / NCLS);
    float d = dis[n];
    vs[i] = d * (y[i] - GAMMA_C * d * t[i]);
}

// xc = y - gamma*dis*t
__global__ void k_compute_xc(float* __restrict__ xc, const float* __restrict__ y,
                             const float* __restrict__ t, const float* __restrict__ dis,
                             size_t total) {
    size_t i = (size_t)blockIdx.x * blockDim.x + threadIdx.x;
    if (i >= total) return;
    int n = (int)(i / NCLS);
    float d = dis[n];
    xc[i] = y[i] - GAMMA_C * d * t[i];
}

// Fused prox + inc_t scatter:
//   zb = z + beta*(vs[src]-vs[dst]);  z = clamp_rownorm(zb, LAMBDA1)
//   t[src] += z_new; t[dst] -= z_new   (t must be pre-zeroed)
__global__ void k_update_z_scatter(float* __restrict__ z, const float* __restrict__ vs,
                                   float* __restrict__ t,
                                   const int* __restrict__ src, const int* __restrict__ dst,
                                   int E) {
    int e = blockIdx.x * blockDim.x + threadIdx.x;
    if (e >= E) return;
    int s = src[e], d = dst[e];
    const float4* ps = (const float4*)(vs + (size_t)s * NCLS);
    const float4* pd = (const float4*)(vs + (size_t)d * NCLS);
    float4* pz = (float4*)(z + (size_t)e * NCLS);
    float4 zb[10];
    float sum = 0.0f;
    #pragma unroll
    for (int q = 0; q < 10; ++q) {
        float4 zv = pz[q], a = ps[q], b = pd[q];
        float4 r;
        r.x = zv.x + BETA_C * (a.x - b.x);
        r.y = zv.y + BETA_C * (a.y - b.y);
        r.z = zv.z + BETA_C * (a.z - b.z);
        r.w = zv.w + BETA_C * (a.w - b.w);
        sum += r.x * r.x + r.y * r.y + r.z * r.z + r.w * r.w;
        zb[q] = r;
    }
    float rn = sqrtf(sum);
    float scale = (rn > 0.0f) ? (fminf(rn, LAM1_C) / rn) : 0.0f;
    float* ts = t + (size_t)s * NCLS;
    float* td = t + (size_t)d * NCLS;
    #pragma unroll
    for (int q = 0; q < 10; ++q) {
        float4 r = zb[q];
        r.x *= scale; r.y *= scale; r.z *= scale; r.w *= scale;
        pz[q] = r;
        atomicAdd(ts + 4*q + 0,  r.x); atomicAdd(ts + 4*q + 1,  r.y);
        atomicAdd(ts + 4*q + 2,  r.z); atomicAdd(ts + 4*q + 3,  r.w);
        atomicAdd(td + 4*q + 0, -r.x); atomicAdd(td + 4*q + 1, -r.y);
        atomicAdd(td + 4*q + 2, -r.z); atomicAdd(td + 4*q + 3, -r.w);
    }
}

// --------------------------------------------------------------- softmax
__global__ void k_logsoftmax(float* __restrict__ out, const float* __restrict__ xc, int N) {
    int n = blockIdx.x * blockDim.x + threadIdx.x;
    if (n >= N) return;
    const float4* p = (const float4*)(xc + (size_t)n * NCLS);
    float v[NCLS];
    float m = -INFINITY;
    #pragma unroll
    for (int q = 0; q < 10; ++q) {
        float4 f = p[q];
        v[4*q+0] = f.x; v[4*q+1] = f.y; v[4*q+2] = f.z; v[4*q+3] = f.w;
    }
    #pragma unroll
    for (int c = 0; c < NCLS; ++c) m = fmaxf(m, v[c]);
    float sum = 0.0f;
    #pragma unroll
    for (int c = 0; c < NCLS; ++c) sum += expf(v[c] - m);
    float l = logf(sum);
    float* o = out + (size_t)n * NCLS;
    #pragma unroll
    for (int c = 0; c < NCLS; ++c) o[c] = v[c] - m - l;
}

// ------------------------------------------------------------------ driver
extern "C" void kernel_launch(void* const* d_in, const int* in_sizes, int n_in,
                              void* d_out, int out_size, void* d_ws, size_t ws_size,
                              hipStream_t stream) {
    const float* x  = (const float*)d_in[0];
    const float* W1 = (const float*)d_in[1];
    const float* W2 = (const float*)d_in[2];
    const int*  src = (const int*)d_in[3];
    const int*  dst = (const int*)d_in[4];
    const int N = in_sizes[0] / F_IN;   // 100000
    const int E = in_sizes[3];          // 1600000
    float* out = (float*)d_out;

    char* w = (char*)d_ws;
    auto take = [&](size_t bytes) {
        char* p = w;
        w += (bytes + 255) & ~(size_t)255;
        return (void*)p;
    };
    __bf16*  w1t = (__bf16*)take((size_t)HID * KPAD * 2);
    __bf16*  w2t = (__bf16*)take((size_t)NCLSPAD * HID * 2);
    __bf16*  hbf = (__bf16*)take((size_t)N * HID * 2);
    float*    hh = (float*)take((size_t)N * NCLS * 4);
    float*    xc = (float*)take((size_t)N * NCLS * 4);
    float*     y = (float*)take((size_t)N * NCLS * 4);
    float*     t = (float*)take((size_t)N * NCLS * 4);
    float*    vs = (float*)take((size_t)N * NCLS * 4);
    float*     z = (float*)take((size_t)E * NCLS * 4);
    unsigned* cnt = (unsigned*)take((size_t)N * 4);
    float*   dis = (float*)take((size_t)N * 4);

    const int B = 256;
    const size_t totNC = (size_t)N * NCLS;
    const size_t zN    = (size_t)E * NCLS;
    const unsigned gNC = (unsigned)((totNC + B - 1) / B);
    const unsigned gE  = (unsigned)((E + B - 1) / B);
    const dim3 gE10(gE, 10);

    // degree -> dis = (deg+1)^-1/2
    k_zero_u32<<<(unsigned)((N + B - 1) / B), B, 0, stream>>>(cnt, (size_t)N);
    k_count<<<gE, B, 0, stream>>>(src, dst, cnt, E);
    k_dis<<<(unsigned)((N + B - 1) / B), B, 0, stream>>>(cnt, dis, N);

    // transposed bf16 weights
    k_prep_w1t<<<(unsigned)((HID * KPAD + B - 1) / B), B, 0, stream>>>(W1, w1t);
    k_prep_w2t<<<(unsigned)((NCLSPAD * HID + B - 1) / B), B, 0, stream>>>(W2, w2t);

    // MLP via WMMA
    k_gemm1<<<(unsigned)(N / 16), 256, 0, stream>>>(x, w1t, hbf);
    k_gemm2<<<(unsigned)(N / 16), 96, 0, stream>>>(hbf, w2t, hh, xc);

    // z = 0 ; t = inc_t-scatter(z=0) = 0
    k_zero_f32<<<(unsigned)((zN + B - 1) / B), B, 0, stream>>>(z, zN);
    k_zero_f32<<<gNC, B, 0, stream>>>(t, totNC);

    for (int it = 0; it < 5; ++it) {
        // y = gamma*hh + (1-gamma)*A_hat@xc
        k_spmm_init<<<gNC, B, 0, stream>>>(y, hh, xc, dis, totNC);
        k_spmm_edges<<<gE10, B, 0, stream>>>(y, xc, dis, src, dst, E);
        // vs = dis * (y - gamma*inc_t(z_prev))   [t holds scatter of z_prev]
        k_compute_vs<<<gNC, B, 0, stream>>>(vs, y, t, dis, totNC);
        // z = prox(z + beta*inc(x_bar)); t = scatter(z_new)
        k_zero_f32<<<gNC, B, 0, stream>>>(t, totNC);
        k_update_z_scatter<<<gE, B, 0, stream>>>(z, vs, t, src, dst, E);
        // xc = y - gamma*inc_t(z_new)
        k_compute_xc<<<gNC, B, 0, stream>>>(xc, y, t, dis, totNC);
    }

    k_logsoftmax<<<(unsigned)((N + B - 1) / B), B, 0, stream>>>(out, xc, N);
}
